// EdgeClassificationGNN_79010218377428
// MI455X (gfx1250) — compile-verified, matrix operand order checked by reading, not run on hardware
//
#include <hip/hip_runtime.h>
#include <hip/hip_bf16.h>

typedef __attribute__((ext_vector_type(16))) _Float16 v16h;
typedef __attribute__((ext_vector_type(8)))  _Float16 v8h;
typedef __attribute__((ext_vector_type(8)))  float    v8f;

// ---------------------------------------------------------------------------
// WMMA GEMM:  Y[M,N] = A[M,Kpad](f16) x Wt[N,Kpad](f16, pre-transposed) + bias
// Block: 256 threads = 8 waves.  Block tile 128x64; each wave owns a 32x32 C
// tile = 4 accumulators -> 4 v_wmma per 32-wide K-step, 8 ds_load_b128.
// Output: f32 (Y) or f16 (Yh) selected by pointer.
// ---------------------------------------------------------------------------
__global__ __launch_bounds__(256)
void k_gemm(const _Float16* __restrict__ A, const _Float16* __restrict__ Wt,
            const float* __restrict__ bias, float* __restrict__ Y,
            _Float16* __restrict__ Yh, int M, int N, int Kpad)
{
    __shared__ _Float16 As[128][32];   // A tile (8 KB)
    __shared__ _Float16 Bs[64][32];    // Wt tile, already [n][k] major (4 KB)

    const int tid  = threadIdx.x;
    const int wave = tid >> 5;
    const int lane = tid & 31;
    const int m0 = blockIdx.x * 128;
    const int n0 = blockIdx.y * 64;
    const int wm = (wave >> 1) * 32;   // 0,32,64,96
    const int wn = (wave & 1) * 32;    // 0,32
    const int hgrp = lane >> 4;
    const int l15  = lane & 15;
    const int off  = hgrp * 8;

    v8f acc00 = {}, acc01 = {}, acc10 = {}, acc11 = {};

    const int ar = tid >> 1;            // A tile row this thread fills (0..127)
    const int ac = (tid & 1) * 16;      // A tile col chunk (0 or 16)
    const int br = tid >> 2;            // B tile row (0..63)
    const int bc = (tid & 3) * 8;       // B tile col chunk

    for (int k0 = 0; k0 < Kpad; k0 += 32) {
        // ---- fill A tile: 128x32 halves, 2 x v8h per thread ----
        {
            int gm = m0 + ar;
            v8h v0 = {}, v1 = {};
            if (gm < M) {
                const _Float16* src = A + (size_t)gm * Kpad + k0 + ac;
                v0 = *(const v8h*)(src);
                v1 = *(const v8h*)(src + 8);
                if (k0 + 32 < Kpad) __builtin_prefetch(src + 32, 0, 1);
            }
            *(v8h*)&As[ar][ac]     = v0;
            *(v8h*)&As[ar][ac + 8] = v1;
        }
        // ---- fill B tile: Wt is [N][Kpad] so this is a straight copy ----
        {
            const _Float16* src = Wt + (size_t)(n0 + br) * Kpad + k0 + bc;
            *(v8h*)&Bs[br][bc] = *(const v8h*)src;
        }
        __syncthreads();

        // ---- fragments (CDNA5 16-bit A/B layout) + 4 WMMAs ----
        v8h a0lo = *(const v8h*)&As[wm + l15][off];
        v8h a0hi = *(const v8h*)&As[wm + l15][16 + off];
        v8h a1lo = *(const v8h*)&As[wm + 16 + l15][off];
        v8h a1hi = *(const v8h*)&As[wm + 16 + l15][16 + off];
        v8h b0lo = *(const v8h*)&Bs[wn + l15][off];
        v8h b0hi = *(const v8h*)&Bs[wn + l15][16 + off];
        v8h b1lo = *(const v8h*)&Bs[wn + 16 + l15][off];
        v8h b1hi = *(const v8h*)&Bs[wn + 16 + l15][16 + off];
        v16h a0 = __builtin_shufflevector(a0lo, a0hi, 0,1,2,3,4,5,6,7,8,9,10,11,12,13,14,15);
        v16h a1 = __builtin_shufflevector(a1lo, a1hi, 0,1,2,3,4,5,6,7,8,9,10,11,12,13,14,15);
        v16h b0 = __builtin_shufflevector(b0lo, b0hi, 0,1,2,3,4,5,6,7,8,9,10,11,12,13,14,15);
        v16h b1 = __builtin_shufflevector(b1lo, b1hi, 0,1,2,3,4,5,6,7,8,9,10,11,12,13,14,15);

        acc00 = __builtin_amdgcn_wmma_f32_16x16x32_f16(false, a0, false, b0, (short)0, acc00, false, false);
        acc01 = __builtin_amdgcn_wmma_f32_16x16x32_f16(false, a0, false, b1, (short)0, acc01, false, false);
        acc10 = __builtin_amdgcn_wmma_f32_16x16x32_f16(false, a1, false, b0, (short)0, acc10, false, false);
        acc11 = __builtin_amdgcn_wmma_f32_16x16x32_f16(false, a1, false, b1, (short)0, acc11, false, false);
        __syncthreads();
    }

    // ---- epilogue: bias add, bounds-guarded store (f32 or f16) ----
#pragma unroll
    for (int mi = 0; mi < 2; ++mi) {
        v8f accs0 = mi ? acc10 : acc00;
        v8f accs1 = mi ? acc11 : acc01;
        const int rowbase = m0 + wm + mi * 16 + hgrp * 8;
#pragma unroll
        for (int ni = 0; ni < 2; ++ni) {
            v8f a = ni ? accs1 : accs0;
            const int colg = n0 + wn + ni * 16 + l15;
            const float bv = bias ? bias[colg] : 0.0f;
#pragma unroll
            for (int r = 0; r < 8; ++r) {
                int rowg = rowbase + r;
                if (rowg < M) {
                    float val = a[r] + bv;
                    if (Yh) Yh[(size_t)rowg * N + colg] = (_Float16)val;
                    else    Y [(size_t)rowg * N + colg] = val;
                }
            }
        }
    }
}

// ---------------------------------------------------------------------------
// Row-wise LayerNorm + tanh.  One block per row, dim <= 256.
// ---------------------------------------------------------------------------
__global__ __launch_bounds__(256)
void k_ln_tanh(const float* __restrict__ Y, const float* __restrict__ g,
               const float* __restrict__ be, _Float16* __restrict__ out, int dim)
{
    __shared__ float red[256];
    const int rowi = blockIdx.x;
    const int t = threadIdx.x;
    const float* y = Y + (size_t)rowi * dim;
    float v = (t < dim) ? y[t] : 0.0f;

    red[t] = v; __syncthreads();
    for (int s = 128; s > 0; s >>= 1) { if (t < s) red[t] += red[t + s]; __syncthreads(); }
    float mean = red[0] / (float)dim; __syncthreads();

    float d = (t < dim) ? (v - mean) : 0.0f;
    red[t] = d * d; __syncthreads();
    for (int s = 128; s > 0; s >>= 1) { if (t < s) red[t] += red[t + s]; __syncthreads(); }
    float var = red[0] / (float)dim;

    if (t < dim) {
        float hn = (v - mean) * rsqrtf(var + 1e-5f) * g[t] + be[t];
        out[(size_t)rowi * dim + t] = (_Float16)tanhf(hn);
    }
}

// ---------------------------------------------------------------------------
// Weight convert f32 -> f16, transposed to [N][Kpad], K padded with zeros.
// ---------------------------------------------------------------------------
__global__ void k_pad_weight_t(const float* __restrict__ W, _Float16* __restrict__ Wt,
                               int K, int N, int Kpad)
{
    int idx = blockIdx.x * blockDim.x + threadIdx.x;
    if (idx >= Kpad * N) return;
    int n = idx / Kpad, k = idx % Kpad;
    Wt[idx] = (k < K) ? (_Float16)W[(size_t)k * N + n] : (_Float16)0.0f;
}

__global__ void k_zero(float* __restrict__ p, int n)
{
    int i = blockIdx.x * blockDim.x + threadIdx.x;
    if (i < n) p[i] = 0.0f;
}

__global__ void k_copy(const float* __restrict__ s, float* __restrict__ d, int n)
{
    int i = blockIdx.x * blockDim.x + threadIdx.x;
    if (i < n) d[i] = s[i];
}

// Final linear with out_dim == 1 (matrix-vector), optional sigmoid.
__global__ void k_linear1(const _Float16* __restrict__ A, int Kpad,
                          const float* __restrict__ W, const float* __restrict__ b,
                          float* __restrict__ out, int M, int K, int sig)
{
    int r = blockIdx.x * blockDim.x + threadIdx.x;
    if (r >= M) return;
    const _Float16* a = A + (size_t)r * Kpad;
    float s = 0.0f;
    for (int k = 0; k < K; ++k) s += (float)a[k] * W[k];
    s += b[0];
    if (sig) s = 1.0f / (1.0f + expf(-s));
    out[r] = s;
}

// ---------------------------------------------------------------------------
// Concat / gather kernels producing padded f16 input matrices
// ---------------------------------------------------------------------------
__global__ void k_concat_edge_enc(const float* __restrict__ x, const float* __restrict__ ea,
                                  const int* __restrict__ row, const int* __restrict__ col,
                                  _Float16* __restrict__ out, int E)
{
    int e = blockIdx.x * blockDim.x + threadIdx.x;
    if (e >= E) return;
    _Float16* o = out + (size_t)e * 32;
    int r = row[e], c = col[e];
    o[0] = (_Float16)x[r * 3 + 0]; o[1] = (_Float16)x[r * 3 + 1]; o[2] = (_Float16)x[r * 3 + 2];
    o[3] = (_Float16)x[c * 3 + 0]; o[4] = (_Float16)x[c * 3 + 1]; o[5] = (_Float16)x[c * 3 + 2];
    o[6] = (_Float16)ea[e * 4 + 0]; o[7] = (_Float16)ea[e * 4 + 1];
    o[8] = (_Float16)ea[e * 4 + 2]; o[9] = (_Float16)ea[e * 4 + 3];
    for (int i = 10; i < 32; ++i) o[i] = (_Float16)0.0f;
}

__global__ void k_concat_node_in(const float* __restrict__ x, _Float16* __restrict__ out, int NN)
{
    int n = blockIdx.x * blockDim.x + threadIdx.x;
    if (n >= NN) return;
    _Float16* o = out + (size_t)n * 32;
    o[0] = (_Float16)x[n * 3 + 0]; o[1] = (_Float16)x[n * 3 + 1]; o[2] = (_Float16)x[n * 3 + 2];
    for (int i = 3; i < 32; ++i) o[i] = (_Float16)0.0f;
}

// [h_row(64), h_col(64), eh(64), (label), x_row(3), x_col(3), ea(4)] pad->224
__global__ void k_concat_ef(const _Float16* __restrict__ h, const _Float16* __restrict__ eh,
                            const float* __restrict__ labels,
                            const float* __restrict__ x, const float* __restrict__ ea,
                            const int* __restrict__ row, const int* __restrict__ col,
                            _Float16* __restrict__ out, int E, int has_label)
{
    int e = blockIdx.x * blockDim.x + threadIdx.x;
    if (e >= E) return;
    _Float16* o = out + (size_t)e * 224;
    int r = row[e], c = col[e];
    int p = 0;
    for (int i = 0; i < 64; ++i) o[p++] = h[(size_t)r * 64 + i];
    for (int i = 0; i < 64; ++i) o[p++] = h[(size_t)c * 64 + i];
    for (int i = 0; i < 64; ++i) o[p++] = eh[(size_t)e * 64 + i];
    if (has_label) o[p++] = (_Float16)labels[e];
    o[p++] = (_Float16)x[r * 3 + 0]; o[p++] = (_Float16)x[r * 3 + 1]; o[p++] = (_Float16)x[r * 3 + 2];
    o[p++] = (_Float16)x[c * 3 + 0]; o[p++] = (_Float16)x[c * 3 + 1]; o[p++] = (_Float16)x[c * 3 + 2];
    o[p++] = (_Float16)ea[e * 4 + 0]; o[p++] = (_Float16)ea[e * 4 + 1];
    o[p++] = (_Float16)ea[e * 4 + 2]; o[p++] = (_Float16)ea[e * 4 + 3];
    while (p < 224) o[p++] = (_Float16)0.0f;
}

// [label, eh(64)] pad->96
__global__ void k_concat_w_in(const float* __restrict__ labels, const _Float16* __restrict__ eh,
                              _Float16* __restrict__ out, int E)
{
    int e = blockIdx.x * blockDim.x + threadIdx.x;
    if (e >= E) return;
    _Float16* o = out + (size_t)e * 96;
    o[0] = (_Float16)labels[e];
    for (int i = 0; i < 64; ++i) o[1 + i] = eh[(size_t)e * 64 + i];
    for (int i = 65; i < 96; ++i) o[i] = (_Float16)0.0f;
}

// [h(64), x(3), one_hop(128), two_hop(128)] pad->352
__global__ void k_concat_x_in(const _Float16* __restrict__ h, const float* __restrict__ x,
                              const float* __restrict__ oh, const float* __restrict__ th,
                              _Float16* __restrict__ out, int NN)
{
    int n = blockIdx.x * blockDim.x + threadIdx.x;
    if (n >= NN) return;
    _Float16* o = out + (size_t)n * 352;
    int p = 0;
    for (int i = 0; i < 64; ++i) o[p++] = h[(size_t)n * 64 + i];
    o[p++] = (_Float16)x[n * 3 + 0]; o[p++] = (_Float16)x[n * 3 + 1]; o[p++] = (_Float16)x[n * 3 + 2];
    for (int i = 0; i < 128; ++i) o[p++] = (_Float16)oh[(size_t)n * 128 + i];
    for (int i = 0; i < 128; ++i) o[p++] = (_Float16)th[(size_t)n * 128 + i];
    while (p < 352) o[p++] = (_Float16)0.0f;
}

// ---------------------------------------------------------------------------
// Weighted message passing (segment_sum via float atomics)
// ---------------------------------------------------------------------------
__global__ void k_msg1(const float* __restrict__ w, const _Float16* __restrict__ h,
                       const _Float16* __restrict__ eh,
                       const int* __restrict__ row, const int* __restrict__ col,
                       float* __restrict__ one_hop, int E)
{
    int idx = blockIdx.x * blockDim.x + threadIdx.x;
    if (idx >= E * 128) return;
    int e = idx >> 7, f = idx & 127;
    float val = (f < 64) ? (float)h[(size_t)row[e] * 64 + f]
                         : (float)eh[(size_t)e * 64 + (f - 64)];
    atomicAdd(&one_hop[(size_t)col[e] * 128 + f], w[e] * val);
}

__global__ void k_msg2(const float* __restrict__ w, const _Float16* __restrict__ h,
                       const _Float16* __restrict__ eh,
                       const int* __restrict__ row, const int* __restrict__ col,
                       const float* __restrict__ one_hop, float* __restrict__ two_hop, int E)
{
    int idx = blockIdx.x * blockDim.x + threadIdx.x;
    if (idx >= E * 128) return;
    int e = idx >> 7, f = idx & 127;
    float we = w[e];
    float base = (f < 64) ? (float)h[(size_t)col[e] * 64 + f]
                          : (float)eh[(size_t)e * 64 + (f - 64)];
    float val = we * (one_hop[(size_t)row[e] * 128 + f] - we * base);
    atomicAdd(&two_hop[(size_t)col[e] * 128 + f], val);
}

// ---------------------------------------------------------------------------
// Host orchestration
// ---------------------------------------------------------------------------
extern "C" void kernel_launch(void* const* d_in, const int* in_sizes, int n_in,
                              void* d_out, int out_size, void* d_ws, size_t ws_size,
                              hipStream_t stream)
{
    const float* x  = (const float*)d_in[0];
    const float* ea = (const float*)d_in[1];
    const int E  = in_sizes[1] / 4;
    const int NN = in_sizes[0] / 3;
    const int* eidx = (const int*)d_in[72];
    const int* row = eidx;
    const int* col = eidx + E;

    struct MLPp { const float* W[3]; const float* bb[3]; const float* g[2]; const float* be[2]; };
    MLPp mp[7];
    int pi = 2;
    for (int m = 0; m < 7; ++m) {
        for (int l = 0; l < 2; ++l) {
            mp[m].W[l]  = (const float*)d_in[pi++];
            mp[m].bb[l] = (const float*)d_in[pi++];
            mp[m].g[l]  = (const float*)d_in[pi++];
            mp[m].be[l] = (const float*)d_in[pi++];
        }
        mp[m].W[2]  = (const float*)d_in[pi++];
        mp[m].bb[2] = (const float*)d_in[pi++];
    }
    // 0 node_enc, 1 edge_enc, 2 init_cls, 3 x_mlp, 4 weight_mlp, 5 edge_mlp, 6 edge_cls
    static const int DIMS[7][4] = {
        {3, 128, 128, 64}, {10, 128, 128, 64}, {202, 128, 128, 1},
        {323, 256, 128, 64}, {65, 128, 128, 1}, {203, 128, 128, 64},
        {203, 128, 128, 1},
    };
    auto al32 = [](int k) { return (k + 31) & ~31; };

    // workspace bump allocator
    char* base = (char*)d_ws;
    size_t off = 0;
    auto alloc = [&](size_t bytes) -> void* {
        off = (off + 255) & ~(size_t)255;
        void* p = base + off;
        off += bytes;
        return p;
    };

    _Float16* Wt[7][3] = {};
    for (int m = 0; m < 7; ++m)
        for (int l = 0; l < 3; ++l) {
            int K = DIMS[m][l], N = DIMS[m][l + 1];
            if (N >= 32) Wt[m][l] = (_Float16*)alloc((size_t)al32(K) * N * sizeof(_Float16));
        }

    _Float16* buf_edge_in = (_Float16*)alloc((size_t)E * 32 * sizeof(_Float16));
    _Float16* buf_node_in = (_Float16*)alloc((size_t)NN * 32 * sizeof(_Float16));
    _Float16* buf_ef      = (_Float16*)alloc((size_t)E * 224 * sizeof(_Float16));
    _Float16* buf_w_in    = (_Float16*)alloc((size_t)E * 96 * sizeof(_Float16));
    _Float16* buf_x_in    = (_Float16*)alloc((size_t)NN * 352 * sizeof(_Float16));
    _Float16* hidA        = (_Float16*)alloc((size_t)E * 256 * sizeof(_Float16));
    float*    Yf32        = (float*)   alloc((size_t)E * 256 * sizeof(float));
    _Float16* h_f16       = (_Float16*)alloc((size_t)NN * 64 * sizeof(_Float16));
    _Float16* eh_f16      = (_Float16*)alloc((size_t)E * 64 * sizeof(_Float16));
    float*    labels      = (float*)   alloc((size_t)E * sizeof(float));
    float*    wbuf        = (float*)   alloc((size_t)E * sizeof(float));
    float*    one_hop     = (float*)   alloc((size_t)NN * 128 * sizeof(float));
    float*    two_hop     = (float*)   alloc((size_t)NN * 128 * sizeof(float));

    auto cdiv = [](int a, int b) { return (a + b - 1) / b; };
    auto gemm = [&](const _Float16* A, const _Float16* B, const float* bias,
                    float* Y, _Float16* Yh, int M, int N, int Kpad) {
        k_gemm<<<dim3(cdiv(M, 128), N / 64), dim3(256), 0, stream>>>(A, B, bias, Y, Yh,
                                                                     M, N, Kpad);
    };
    auto ln = [&](const float* Y, const float* g, const float* be, _Float16* out, int M, int dim) {
        k_ln_tanh<<<dim3(M), dim3(256), 0, stream>>>(Y, g, be, out, dim);
    };
    auto hidden2 = [&](int mi, const _Float16* A0, int M) {
        gemm(A0, Wt[mi][0], mp[mi].bb[0], Yf32, nullptr, M, DIMS[mi][1], al32(DIMS[mi][0]));
        ln(Yf32, mp[mi].g[0], mp[mi].be[0], hidA, M, DIMS[mi][1]);
        gemm(hidA, Wt[mi][1], mp[mi].bb[1], Yf32, nullptr, M, DIMS[mi][2], DIMS[mi][1]);
        ln(Yf32, mp[mi].g[1], mp[mi].be[1], hidA, M, DIMS[mi][2]);
    };
    auto final64 = [&](int mi, _Float16* dst, int M) {
        gemm(hidA, Wt[mi][2], mp[mi].bb[2], nullptr, dst, M, 64, 128);
    };
    auto final1 = [&](int mi, float* dst, int M, int sig) {
        k_linear1<<<dim3(cdiv(M, 256)), dim3(256), 0, stream>>>(hidA, 128, mp[mi].W[2],
                                                                mp[mi].bb[2], dst, M, 128, sig);
    };

    // ---- convert all WMMA weights to transposed padded f16 ----
    for (int m = 0; m < 7; ++m)
        for (int l = 0; l < 3; ++l) {
            if (!Wt[m][l]) continue;
            int K = DIMS[m][l], N = DIMS[m][l + 1], Kp = al32(K);
            k_pad_weight_t<<<dim3(cdiv(Kp * N, 256)), dim3(256), 0, stream>>>(
                mp[m].W[l], Wt[m][l], K, N, Kp);
        }

    const int gE = cdiv(E, 256), gN = cdiv(NN, 256), gEF = cdiv(E * 128, 256);

    // ---- encoders ----
    k_concat_edge_enc<<<dim3(gE), dim3(256), 0, stream>>>(x, ea, row, col, buf_edge_in, E);
    hidden2(1, buf_edge_in, E);
    final64(1, eh_f16, E);

    k_concat_node_in<<<dim3(gN), dim3(256), 0, stream>>>(x, buf_node_in, NN);
    hidden2(0, buf_node_in, NN);
    final64(0, h_f16, NN);

    // ---- initial classifier ----
    k_concat_ef<<<dim3(gE), dim3(256), 0, stream>>>(h_f16, eh_f16, nullptr, x, ea, row, col,
                                                    buf_ef, E, 0);
    hidden2(2, buf_ef, E);
    final1(2, labels, E, 1);

    // ---- message-passing iterations ----
    for (int it = 0; it < 6; ++it) {
        // edge weights
        k_concat_w_in<<<dim3(gE), dim3(256), 0, stream>>>(labels, eh_f16, buf_w_in, E);
        hidden2(4, buf_w_in, E);
        final1(4, wbuf, E, 0);

        // one-hop / two-hop aggregation
        k_zero<<<dim3(cdiv(NN * 128, 256)), dim3(256), 0, stream>>>(one_hop, NN * 128);
        k_msg1<<<dim3(gEF), dim3(256), 0, stream>>>(wbuf, h_f16, eh_f16, row, col, one_hop, E);
        k_zero<<<dim3(cdiv(NN * 128, 256)), dim3(256), 0, stream>>>(two_hop, NN * 128);
        k_msg2<<<dim3(gEF), dim3(256), 0, stream>>>(wbuf, h_f16, eh_f16, row, col, one_hop,
                                                    two_hop, E);

        // node update
        k_concat_x_in<<<dim3(gN), dim3(256), 0, stream>>>(h_f16, x, one_hop, two_hop,
                                                          buf_x_in, NN);
        hidden2(3, buf_x_in, NN);
        final64(3, h_f16, NN);

        // edge feature update
        k_concat_ef<<<dim3(gE), dim3(256), 0, stream>>>(h_f16, eh_f16, labels, x, ea, row, col,
                                                        buf_ef, E, 1);
        hidden2(5, buf_ef, E);
        final64(5, eh_f16, E);

        // edge classification
        k_concat_ef<<<dim3(gE), dim3(256), 0, stream>>>(h_f16, eh_f16, labels, x, ea, row, col,
                                                        buf_ef, E, 1);
        hidden2(6, buf_ef, E);
        final1(6, labels, E, 1);
    }

    k_copy<<<dim3(gE), dim3(256), 0, stream>>>(labels, (float*)d_out, E);
}
